// GroupCrossAttention_6674379178646
// MI455X (gfx1250) — matrix operand order repaired and checked by reading, not compiled, not run on hardware
//
#include <hip/hip_runtime.h>

// ---------- constants (from reference) ----------
#define S_   128
#define H_   2048
#define NH_  16
#define HD_  128
#define K_   16
#define DP_  16
#define SK_  128
#define EPS_ 1e-6f
#define QSCALE 0.08838834764831845f   // 1/sqrt(128)

// ---------- POD vector types ----------
typedef __attribute__((ext_vector_type(16))) __bf16       v16bf;
typedef __attribute__((ext_vector_type(8)))  float        v8f;
typedef __attribute__((ext_vector_type(4)))  unsigned int u32x4;
typedef __attribute__((ext_vector_type(2)))  unsigned int u32x2;
typedef __attribute__((ext_vector_type(4)))  float        f32x4;

union FragU { u32x4 u[2]; v16bf v; };

__device__ __forceinline__ v8f wmma_bf16(const FragU& a, const FragU& b, v8f c) {
  return __builtin_amdgcn_wmma_f32_16x16x32_bf16(false, a.v, false, b.v,
                                                 (short)0, c, false, false);
}

// =======================================================================
// Kernel 1: RMSNorm rows for d = 1..16  ->  xn bf16 (2048 x 2048)
// =======================================================================
__global__ __launch_bounds__(256) void k_rmsnorm(const float* __restrict__ hs,
                                                 const float* __restrict__ rw,
                                                 __bf16* __restrict__ xn) {
  int row = blockIdx.x;                 // (d-1)*128 + s, 0..2047
  int d = (row >> 7) + 1, s = row & 127;
  const float* src = hs + ((size_t)(d * S_ + s)) * H_;
  __shared__ float red[256];
  int tid = threadIdx.x;
  const f32x4* s4 = (const f32x4*)src;
  f32x4 vbuf[2];
  float ss = 0.f;
#pragma unroll
  for (int i = 0; i < 2; ++i) {
    f32x4 v = s4[tid * 2 + i];
    vbuf[i] = v;
    ss += v.x * v.x + v.y * v.y + v.z * v.z + v.w * v.w;
  }
  red[tid] = ss; __syncthreads();
  for (int o = 128; o > 0; o >>= 1) {
    if (tid < o) red[tid] += red[tid + o];
    __syncthreads();
  }
  float rn = rsqrtf(red[0] * (1.0f / (float)H_) + EPS_);
  const f32x4* w4 = (const f32x4*)rw;
  __bf16* dst = xn + (size_t)row * H_ + tid * 8;
#pragma unroll
  for (int i = 0; i < 2; ++i) {
    f32x4 v = vbuf[i]; f32x4 w = w4[tid * 2 + i];
    alignas(8) __bf16 p4[4];
    p4[0] = (__bf16)(v.x * rn * w.x);
    p4[1] = (__bf16)(v.y * rn * w.y);
    p4[2] = (__bf16)(v.z * rn * w.z);
    p4[3] = (__bf16)(v.w * rn * w.w);
    *(u32x2*)&dst[i * 4] = *(u32x2*)p4;
  }
}

// =======================================================================
// Kernel 2: fp32 -> bf16 convert (weights)
// =======================================================================
__global__ void k_cvt_bf16(const float* __restrict__ src,
                           __bf16* __restrict__ dst, int n4) {
  int i = blockIdx.x * blockDim.x + threadIdx.x;
  if (i < n4) {
    f32x4 v = *(const f32x4*)(src + (size_t)i * 4);
    alignas(8) __bf16 p4[4];
    p4[0] = (__bf16)v.x; p4[1] = (__bf16)v.y;
    p4[2] = (__bf16)v.z; p4[3] = (__bf16)v.w;
    *(u32x2*)&dst[(size_t)i * 4] = *(u32x2*)p4;
  }
}

// =======================================================================
// Kernel 3/6: 2048x2048x2048 GEMM  C = A * B^T   (A: MxK bf16 row-major,
// B: NxK bf16 row-major). mode 0: store scaled bf16 Q in (d,h,s,e) layout.
// mode 1: store fp32 row-major into outF.
// 128x128 tile per block, 8 waves, each wave: 16 rows x 128 cols (8 WMMA tiles)
// =======================================================================
__global__ __launch_bounds__(256) void k_gemm_bf16(const __bf16* __restrict__ A,
                                                   const __bf16* __restrict__ B,
                                                   __bf16* __restrict__ outQ,
                                                   float* __restrict__ outF,
                                                   int mode) {
  __shared__ alignas(16) __bf16 As[128 * 32];   // [m][k]
  __shared__ alignas(16) __bf16 Bs[32 * 128];   // [n][k]
  const int Kdim = H_;
  int bm = blockIdx.y, bn = blockIdx.x;
  int tid = threadIdx.x, wave = tid >> 5, lane = tid & 31;
  int arow = wave * 16 + (lane & 15);
  int ak = (lane >> 4) * 8;
  v8f acc[8] = {};
  for (int k0 = 0; k0 < Kdim; k0 += 32) {
    {   // A panel: 128x32, each thread 16 contiguous bf16
      int row = tid >> 1, colh = (tid & 1) * 16;
      const __bf16* src = A + (size_t)(bm * 128 + row) * Kdim + k0 + colh;
      *(u32x4*)&As[row * 32 + colh]     = *(const u32x4*)src;
      *(u32x4*)&As[row * 32 + colh + 8] = *(const u32x4*)(src + 8);
    }
    {   // B panel: Bs[n][k], each thread 16 contiguous bf16
      int row = tid >> 1, colh = (tid & 1) * 16;
      const __bf16* src = B + (size_t)(bn * 128 + row) * Kdim + k0 + colh;
      *(u32x4*)&Bs[row * 32 + colh]     = *(const u32x4*)src;
      *(u32x4*)&Bs[row * 32 + colh + 8] = *(const u32x4*)(src + 8);
    }
    __syncthreads();
    FragU a;
    a.u[0] = *(const u32x4*)&As[arow * 32 + ak];
    a.u[1] = *(const u32x4*)&As[arow * 32 + ak + 16];
#pragma unroll
    for (int nt = 0; nt < 8; ++nt) {
      FragU b;   // lane = N, table-order K
      b.u[0] = *(const u32x4*)&Bs[(nt * 16 + (lane & 15)) * 32 + ak];
      b.u[1] = *(const u32x4*)&Bs[(nt * 16 + (lane & 15)) * 32 + ak + 16];
      acc[nt] = wmma_bf16(a, b, acc[nt]);
    }
    __syncthreads();
  }
  // store: C tile (j,lane) -> M = j + (lane>>4)*8, N = lane&15
#pragma unroll
  for (int nt = 0; nt < 8; ++nt) {
#pragma unroll
    for (int j = 0; j < 8; ++j) {
      int m = bm * 128 + wave * 16 + j + ((lane >> 4) << 3);
      int c = bn * 128 + nt * 16 + (lane & 15);
      if (mode == 0) {
        int dd = m >> 7, s = m & 127, hh = c >> 7, e = c & 127;
        outQ[(((size_t)dd * NH_ + hh) * S_ + s) * HD_ + e] =
            (__bf16)(acc[nt][j] * QSCALE);
      } else {
        outF[(size_t)m * H_ + c] = acc[nt][j];
      }
    }
  }
}

// =======================================================================
// Kernel 4: attention. One block per (d, head). 8 waves.
//   Q frags in registers (reused across 16 chunks).
//   KV buffer: K as [t][e] (scores B-operand), then V as [e][t].
//   softmax in registers, chunk weight folded into P (bf16 in LDS).
//   O accumulated across chunks in WMMA C/D registers.
// =======================================================================
__global__ __launch_bounds__(256) void k_attn(const __bf16* __restrict__ qb,
                                              const float* __restrict__ ck,
                                              const float* __restrict__ cv,
                                              const float* __restrict__ wts,
                                              __bf16* __restrict__ wvb) {
  extern __shared__ char smem[];
  __bf16* KV = (__bf16*)smem;                  // 128*128 bf16 = 32 KB
  __bf16* Ps = (__bf16*)(smem + 32768);        // 128*128 bf16 = 32 KB
  int dd = blockIdx.x >> 4;                    // 0..15 (d-1)
  int h  = blockIdx.x & 15;
  int tid = threadIdx.x, wave = tid >> 5, lane = tid & 31;
  int arow = wave * 16 + (lane & 15);
  int ak = (lane >> 4) * 8;

  // Q fragments for the wave's 16-row strip, 4 k-steps over e
  FragU qf[4];
  const __bf16* qbase = qb + ((size_t)(dd * NH_ + h) * S_) * HD_;
#pragma unroll
  for (int ke = 0; ke < 4; ++ke) {
    const __bf16* p = qbase + (size_t)arow * HD_ + ke * 32 + ak;
    qf[ke].u[0] = *(const u32x4*)p;
    qf[ke].u[1] = *(const u32x4*)(p + 16);
  }

  v8f oacc[8] = {};

  for (int kk = 0; kk < K_; ++kk) {
    float w = wts[dd * K_ + kk];
    size_t cbase = ((size_t)(dd * K_ + kk)) * SK_ * NH_ * HD_;
    if (kk + 1 < K_) {  // hint next chunk toward L2
      size_t nbase = ((size_t)(dd * K_ + kk + 1)) * SK_ * NH_ * HD_;
      __builtin_prefetch(ck + nbase + ((size_t)(tid >> 1) * NH_ + h) * HD_, 0, 1);
    }
    {   // K chunk fp32 -> bf16, direct: KV[t][e], 8 bf16 per b128 store
      int t = tid >> 1;
      int e0 = (tid & 1) * 64;
      const float* src = ck + cbase + ((size_t)t * NH_ + h) * HD_ + e0;
#pragma unroll
      for (int i = 0; i < 8; ++i) {
        f32x4 fa = *(const f32x4*)(src + i * 8);
        f32x4 fb = *(const f32x4*)(src + i * 8 + 4);
        alignas(16) __bf16 p8[8];
        p8[0] = (__bf16)fa.x; p8[1] = (__bf16)fa.y;
        p8[2] = (__bf16)fa.z; p8[3] = (__bf16)fa.w;
        p8[4] = (__bf16)fb.x; p8[5] = (__bf16)fb.y;
        p8[6] = (__bf16)fb.z; p8[7] = (__bf16)fb.w;
        *(u32x4*)&KV[t * 128 + e0 + i * 8] = *(u32x4*)p8;
      }
    }
    __syncthreads();

    // scores: S(128) x T(128), K-dim = e (128 = 4 steps of 32)
    v8f sc[8] = {};
#pragma unroll
    for (int ke = 0; ke < 4; ++ke) {
#pragma unroll
      for (int tt = 0; tt < 8; ++tt) {
        FragU b;
        const __bf16* kp = &KV[(tt * 16 + (lane & 15)) * 128 + ke * 32 + ak];
        b.u[0] = *(const u32x4*)kp;
        b.u[1] = *(const u32x4*)(kp + 16);
        sc[tt] = wmma_bf16(qf[ke], b, sc[tt]);
      }
    }
    __syncthreads();   // KV reads done; safe to overwrite with V

    // row softmax over t (row j lives across 16 lanes of a half-wave)
#pragma unroll
    for (int j = 0; j < 8; ++j) {
      float mx = -3.0e38f;
#pragma unroll
      for (int tt = 0; tt < 8; ++tt) mx = fmaxf(mx, sc[tt][j]);
#pragma unroll
      for (int m = 1; m <= 8; m <<= 1) mx = fmaxf(mx, __shfl_xor(mx, m, 32));
      float sum = 0.f;
#pragma unroll
      for (int tt = 0; tt < 8; ++tt) {
        float p = __expf(sc[tt][j] - mx);
        sc[tt][j] = p; sum += p;
      }
#pragma unroll
      for (int m = 1; m <= 8; m <<= 1) sum += __shfl_xor(sum, m, 32);
      float r = w / sum;   // fold chunk weight into P
#pragma unroll
      for (int tt = 0; tt < 8; ++tt) sc[tt][j] *= r;
    }
    // write P (bf16) to Ps[s][t]
#pragma unroll
    for (int tt = 0; tt < 8; ++tt) {
#pragma unroll
      for (int j = 0; j < 8; ++j) {
        int srow = wave * 16 + j + ((lane >> 4) << 3);
        Ps[srow * 128 + tt * 16 + (lane & 15)] = (__bf16)sc[tt][j];
      }
    }
    {   // V chunk fp32 -> bf16, transposed: KV[e][t].
        // Thread owns 4t x 16e block (two 8e halves); packs 4 t-values
        // per b64 LDS store.
      int t0 = (tid & 31) * 4;
      int e0b = (tid >> 5) * 16;
#pragma unroll
      for (int half = 0; half < 2; ++half) {
        int eb = e0b + half * 8;
        f32x4 r[4][2];
#pragma unroll
        for (int ti = 0; ti < 4; ++ti) {
          const float* src =
              cv + cbase + ((size_t)(t0 + ti) * NH_ + h) * HD_ + eb;
          r[ti][0] = *(const f32x4*)src;
          r[ti][1] = *(const f32x4*)(src + 4);
        }
#pragma unroll
        for (int c = 0; c < 2; ++c) {
#pragma unroll
          for (int sub = 0; sub < 4; ++sub) {
            int e = eb + c * 4 + sub;
            alignas(8) __bf16 p4[4];
#pragma unroll
            for (int ti = 0; ti < 4; ++ti) p4[ti] = (__bf16)r[ti][c][sub];
            *(u32x2*)&KV[e * 128 + t0] = *(u32x2*)p4;
          }
        }
      }
    }
    __syncthreads();

    // O += P * V  (K-dim = t, 4 steps of 32), accumulates across chunks
#pragma unroll
    for (int kt = 0; kt < 4; ++kt) {
      FragU a;
      const __bf16* p = &Ps[arow * 128 + kt * 32 + ak];
      a.u[0] = *(const u32x4*)p;
      a.u[1] = *(const u32x4*)(p + 16);
#pragma unroll
      for (int et = 0; et < 8; ++et) {
        FragU b;
        const __bf16* vp = &KV[(et * 16 + (lane & 15)) * 128 + kt * 32 + ak];
        b.u[0] = *(const u32x4*)vp;
        b.u[1] = *(const u32x4*)(vp + 16);
        oacc[et] = wmma_bf16(a, b, oacc[et]);
      }
    }
    __syncthreads();
  }
  // store wv bf16: row = dd*128 + s, col = h*128 + e
#pragma unroll
  for (int et = 0; et < 8; ++et) {
#pragma unroll
    for (int j = 0; j < 8; ++j) {
      int s = wave * 16 + j + ((lane >> 4) << 3);
      int e = et * 16 + (lane & 15);
      wvb[((size_t)(dd * S_ + s)) * H_ + h * HD_ + e] = (__bf16)oacc[et][j];
    }
  }
}

// =======================================================================
// Kernel 5: zero first S rows of the output
// =======================================================================
__global__ void k_zero(float* __restrict__ p, int n) {
  int i = blockIdx.x * blockDim.x + threadIdx.x;
  if (i < n) p[i] = 0.f;
}

// =======================================================================
// launch
// =======================================================================
extern "C" void kernel_launch(void* const* d_in, const int* in_sizes, int n_in,
                              void* d_out, int out_size, void* d_ws, size_t ws_size,
                              hipStream_t stream) {
  (void)in_sizes; (void)n_in; (void)out_size; (void)ws_size;
  const float* hs   = (const float*)d_in[0];
  const float* ck   = (const float*)d_in[1];
  const float* cv   = (const float*)d_in[2];
  const float* wts  = (const float*)d_in[3];
  const float* rmsw = (const float*)d_in[4];
  const float* Wq   = (const float*)d_in[5];
  const float* Wo   = (const float*)d_in[6];
  float* out = (float*)d_out;

  const size_t M = 2048;
  __bf16* xn  = (__bf16*)d_ws;        // 2048x2048 bf16 (8 MB)
  __bf16* wqb = xn  + M * M;
  __bf16* wob = wqb + M * M;
  __bf16* qbq = wob + M * M;          // scaled Q, (d,h,s,e) layout
  __bf16* wvb = qbq + M * M;          // attention output, (d*s, h*e) layout

  // 1) RMSNorm -> xn
  k_rmsnorm<<<2048, 256, 0, stream>>>(hs, rmsw, xn);
  // 2) weight converts
  k_cvt_bf16<<<4096, 256, 0, stream>>>(Wq, wqb, 1024 * 1024);
  k_cvt_bf16<<<4096, 256, 0, stream>>>(Wo, wob, 1024 * 1024);
  // 3) Q = xn * Wq^T  (scaled, scattered layout)
  k_gemm_bf16<<<dim3(16, 16), 256, 0, stream>>>(xn, wqb, qbq, nullptr, 0);
  // 4) attention
  k_attn<<<256, 256, 65536, stream>>>(qbq, ck, cv, wts, wvb);
  // 5) zero first S rows of output
  k_zero<<<1024, 256, 0, stream>>>(out, S_ * H_);
  // 6) out[S: ] = wv * Wo^T
  k_gemm_bf16<<<dim3(16, 16), 256, 0, stream>>>(wvb, wob, nullptr,
                                                out + (size_t)S_ * H_, 1);
}
